// RNN_49125835932082
// MI455X (gfx1250) — matrix operand order repaired and checked by reading, not compiled, not run on hardware
//
#include <hip/hip_runtime.h>

// ---- problem sizes (from reference) ----
#define BB 4
#define SS 4096
#define DD 2048
#define NN 32
#define HH 64
#define CC (NN * HH)   // 2048
#define KK 4
#define BS (BB * SS)   // 16384 rows
#define C2 (2 * CC)    // 4096

// ---- vector types for WMMA fragments ----
typedef __attribute__((ext_vector_type(16))) __bf16        v16bf;
typedef __attribute__((ext_vector_type(8)))  float         v8f;
typedef __attribute__((ext_vector_type(2)))  float         v2f;
typedef __attribute__((ext_vector_type(8)))  unsigned int  v8u;
typedef __attribute__((ext_vector_type(4)))  unsigned int  v4u;

#if defined(__has_builtin)
#if __has_builtin(__builtin_amdgcn_wmma_f32_16x16x4_f32)
#define HAVE_WMMA_F32X4 1
#else
#define HAVE_WMMA_F32X4 0
#endif
#else
#define HAVE_WMMA_F32X4 0
#endif

__device__ __forceinline__ unsigned short f32_to_bf16(float f) {
  union { float f; unsigned int u; } v; v.f = f;
  unsigned int u = v.u;
  u += 0x7FFFu + ((u >> 16) & 1u);   // round to nearest even
  return (unsigned short)(u >> 16);
}

// ---------------------------------------------------------------------------
// Kernel 0: f32 -> bf16 conversion (feeds the WMMA GEMMs)
// ---------------------------------------------------------------------------
__global__ void cvt_f32_bf16(const float* __restrict__ in,
                             unsigned short* __restrict__ out, int n) {
  int i = blockIdx.x * blockDim.x + threadIdx.x;
  if (i < n) out[i] = f32_to_bf16(in[i]);
}

// ---------------------------------------------------------------------------
// Kernel 1: GEMM  out[M,N] = A[M,K=2048](bf16) * W[N,K=2048](bf16)^T, f32 acc.
// 256 threads = 8 waves laid out 4 (M) x 2 (N); each wave owns a 32x64 output
// strip: 2 A fragments x 4 B tiles = 8 accumulators, so every B fragment is
// reused by two WMMAs straight from registers (8 WMMA : 12 b128-loads per
// k-step). No LDS staging: W (<=16MB) and the streaming A panel are
// L2-resident on MI455X (192MB L2), so cross-wave reuse is serviced by L2.
// K is a compile-time constant -> single steady-state loop, no remainder.
//
// Fragment layouts per CDNA5 ISA 7.12.2 (wave32):
//   A (16x32 bf16): lane L holds row M=L%16; VGPR v, half p -> K =
//                   (v/4)*16 + (L/16)*8 + (v%4)*2 + p   => two 16B loads/lane
//   B (32x16 bf16): lane L holds col N=L%16, K same as A:
//                   (e/8)*16 + (L/16)*8 + e%8  (2x16B loads)
//   C/D (f32 16x16): lane L holds col N=L%16, row M=(L/16)*8+v
// ---------------------------------------------------------------------------
__global__ __launch_bounds__(256)
void gemm_bf16_wmma(const unsigned short* __restrict__ A,
                    const unsigned short* __restrict__ W,
                    float* __restrict__ out, int M, int N) {
  const int lane   = threadIdx.x & 31;
  const int wave   = threadIdx.x >> 5;
  const int laneLo = lane & 15;
  const int laneHi = lane >> 4;
  const int mBase  = blockIdx.x * 128 + (wave & 3) * 32;
  const int nBase  = blockIdx.y * 128 + (wave >> 2) * 64;

  v8f acc[2][4] = {};

  const unsigned short* arow0 = A + (size_t)(mBase + laneLo) * DD + laneHi * 8;
  const unsigned short* arow1 = arow0 + (size_t)16 * DD;
  const unsigned short* brows[4];
#pragma unroll
  for (int t = 0; t < 4; ++t)
    brows[t] = W + (size_t)(nBase + t * 16 + laneLo) * DD + laneHi * 8;

#pragma unroll 2
  for (int k0 = 0; k0 < DD; k0 += 32) {
    v4u a0lo = *(const v4u*)(arow0 + k0);
    v4u a0hi = *(const v4u*)(arow0 + k0 + 16);
    v4u a1lo = *(const v4u*)(arow1 + k0);
    v4u a1hi = *(const v4u*)(arow1 + k0 + 16);
    v8u au0, au1;
#pragma unroll
    for (int e = 0; e < 4; ++e) {
      au0[e] = a0lo[e]; au0[e + 4] = a0hi[e];
      au1[e] = a1lo[e]; au1[e + 4] = a1hi[e];
    }
    v16bf a0 = __builtin_bit_cast(v16bf, au0);
    v16bf a1 = __builtin_bit_cast(v16bf, au1);
#pragma unroll
    for (int t = 0; t < 4; ++t) {
      v4u blo = *(const v4u*)(brows[t] + k0);
      v4u bhi = *(const v4u*)(brows[t] + k0 + 16);
      v8u bu;
      for (int e = 0; e < 4; ++e) { bu[e] = blo[e]; bu[e + 4] = bhi[e]; }
      v16bf b = __builtin_bit_cast(v16bf, bu);
      acc[0][t] = __builtin_amdgcn_wmma_f32_16x16x32_bf16(
          false, a0, false, b, (short)0, acc[0][t], false, false);
      acc[1][t] = __builtin_amdgcn_wmma_f32_16x16x32_bf16(
          false, a1, false, b, (short)0, acc[1][t], false, false);
    }
  }

#pragma unroll
  for (int h = 0; h < 2; ++h) {
#pragma unroll
    for (int t = 0; t < 4; ++t) {
      float* o = out + (size_t)(mBase + h * 16 + laneHi * 8) * N
                     + nBase + t * 16 + laneLo;
#pragma unroll
      for (int v = 0; v < 8; ++v) o[(size_t)v * N] = acc[h][t][v];
    }
  }
}

// ---------------------------------------------------------------------------
// Kernel 2: causal depthwise conv (K=4) + SiLU.  Reads xin = xg[:, 0:C].
// ---------------------------------------------------------------------------
__global__ void conv_silu_kernel(const float* __restrict__ xg,
                                 const float* __restrict__ cw,
                                 float* __restrict__ xc) {
  long i = (long)blockIdx.x * blockDim.x + threadIdx.x;
  if (i >= (long)BS * CC) return;
  int  c   = (int)(i % CC);
  long row = i / CC;            // row = b*S + s
  int  s   = (int)(row % SS);
  float acc = 0.f;
#pragma unroll
  for (int j = 0; j < KK; ++j) {
    int sj = s - (KK - 1) + j;
    if (sj >= 0) acc += cw[c * KK + j] * xg[(row + (long)(sj - s)) * C2 + c];
  }
  float sig = 1.f / (1.f + __expf(-acc));
  xc[i] = acc * sig;
}

// ---------------------------------------------------------------------------
// Kernel 3: sequential RNN scan, one wave per head n (32 single-wave blocks).
// Batches b=0..3 as rows 0..3 of a zero-padded 16x64 state matrix; the
// recurrence h = tanh(x + h @ W_n) uses f32 WMMA (V_WMMA_F32_16X16X4_F32,
// confirmed lowering last round) to keep the 4096-step chain in full
// precision. W_n (16KB f32) is staged in LDS.
//   A (16x4 f32): lane L row M=L%16, VGPR v -> K = (L/16)*2 + v
//   B (4x16 f32): lane L col N=L%16, VGPR v -> K = (L/16)*2 + v
// ---------------------------------------------------------------------------
__global__ __launch_bounds__(32)
void rnn_scan_kernel(const float* __restrict__ xc,
                     const float* __restrict__ sw,
                     float* __restrict__ ys) {
  __shared__ float lw[HH * HH];     // W_n, 16KB
  __shared__ float hbuf[16][HH];    // padded state rows (rows 4..15 stay 0)

  const int n    = blockIdx.x;
  const int lane = threadIdx.x;
  const int laneLo = lane & 15;
  const int laneHi = lane >> 4;

  for (int i = lane; i < HH * HH; i += 32) lw[i] = sw[(size_t)n * HH * HH + i];
  for (int i = lane; i < 16 * HH; i += 32) (&hbuf[0][0])[i] = 0.f;
  __syncthreads();

  for (int s = 0; s < SS; ++s) {
#if HAVE_WMMA_F32X4
    v8f acc[4];
#pragma unroll
    for (int t = 0; t < 4; ++t) {
#pragma unroll
      for (int v = 0; v < 8; ++v) {
        int m = laneHi * 8 + v;
        acc[t][v] = (m < BB)
            ? xc[(((size_t)m * SS + s) * CC) + n * HH + t * 16 + laneLo]
            : 0.f;
      }
    }
#pragma unroll 4
    for (int kt = 0; kt < 16; ++kt) {
      int kb = kt * 4 + laneHi * 2;
      v2f a;
      a[0] = hbuf[laneLo][kb];
      a[1] = hbuf[laneLo][kb + 1];
#pragma unroll
      for (int t = 0; t < 4; ++t) {
        v2f b;
        b[0] = lw[(kb)     * HH + t * 16 + laneLo];
        b[1] = lw[(kb + 1) * HH + t * 16 + laneLo];
        acc[t] = __builtin_amdgcn_wmma_f32_16x16x4_f32(
            false, a, false, b, (short)0, acc[t], false, false);
      }
    }
    __syncthreads();
#pragma unroll
    for (int t = 0; t < 4; ++t) {
#pragma unroll
      for (int v = 0; v < 8; ++v) {
        int m = laneHi * 8 + v;
        if (m < BB) {
          float h = tanhf(acc[t][v]);
          hbuf[m][t * 16 + laneLo] = h;
          ys[(((size_t)m * SS + s) * CC) + n * HH + t * 16 + laneLo] = h;
        }
      }
    }
    __syncthreads();
#else
    // VALU fallback: each lane computes 8 (b,k) outputs.
    float outv[8];
#pragma unroll
    for (int i = 0; i < 8; ++i) {
      int o = lane * 8 + i;          // 0..255
      int b = o >> 6, k = o & 63;
      float acc = xc[(((size_t)b * SS + s) * CC) + n * HH + k];
      for (int h = 0; h < HH; ++h) acc += hbuf[b][h] * lw[h * HH + k];
      outv[i] = tanhf(acc);
    }
    __syncthreads();
#pragma unroll
    for (int i = 0; i < 8; ++i) {
      int o = lane * 8 + i;
      int b = o >> 6, k = o & 63;
      hbuf[b][k] = outv[i];
      ys[(((size_t)b * SS + s) * CC) + n * HH + k] = outv[i];
    }
    __syncthreads();
#endif
  }
}

// ---------------------------------------------------------------------------
// Kernel 4: y = rmsnorm(y * silu(g)) * norm_w, emitted as bf16 for GEMM2.
// One 256-thread block per (b,s) row of C=2048 elements.
// ---------------------------------------------------------------------------
__global__ __launch_bounds__(256)
void gate_norm_kernel(const float* __restrict__ ys,
                      const float* __restrict__ xg,
                      const float* __restrict__ nw,
                      unsigned short* __restrict__ yb) {
  __shared__ float red[8];
  const long row = blockIdx.x;
  const int  t   = threadIdx.x;
  const int  lane = t & 31, wave = t >> 5;

  float vals[8];
  float ss = 0.f;
#pragma unroll
  for (int i = 0; i < 8; ++i) {
    int c = t + i * 256;
    float y = ys[row * CC + c];
    float g = xg[row * C2 + CC + c];
    float v = y * (g / (1.f + __expf(-g)));   // y * silu(g)
    vals[i] = v;
    ss += v * v;
  }
  for (int o = 16; o > 0; o >>= 1) ss += __shfl_xor(ss, o);   // wave32 tree
  if (lane == 0) red[wave] = ss;
  __syncthreads();
  if (t == 0) {
    float tot = 0.f;
    for (int w = 0; w < 8; ++w) tot += red[w];
    red[0] = rsqrtf(tot / (float)CC + 1e-6f);
  }
  __syncthreads();
  float rs = red[0];
#pragma unroll
  for (int i = 0; i < 8; ++i) {
    int c = t + i * 256;
    yb[row * CC + c] = f32_to_bf16(vals[i] * rs * nw[c]);
  }
}

// ---------------------------------------------------------------------------
// Workspace layout (bytes):
//   xb  (bf16 x)        @ 0           64 MB
//   wb  (bf16 w_in)     @ 67108864    16 MB
//   wob (bf16 w_out)    @ 83886080     8 MB
//   xg  (f32, BSx2C)    @ 92274688   256 MB
//   xc  (f32, BSxC)     @ 360710144  128 MB
//   ys  (f32, BSxC)     @ 494927872  128 MB
//   yb  (bf16, BSxC)    @ 629145600   64 MB     total ~664 MB
// ---------------------------------------------------------------------------
extern "C" void kernel_launch(void* const* d_in, const int* in_sizes, int n_in,
                              void* d_out, int out_size, void* d_ws, size_t ws_size,
                              hipStream_t stream) {
  const float* x       = (const float*)d_in[0];
  const float* w_in    = (const float*)d_in[1];
  const float* conv_w  = (const float*)d_in[2];
  const float* state_w = (const float*)d_in[3];
  const float* norm_w  = (const float*)d_in[4];
  const float* w_out   = (const float*)d_in[5];
  float* out = (float*)d_out;

  char* ws = (char*)d_ws;
  unsigned short* xb  = (unsigned short*)(ws);
  unsigned short* wb  = (unsigned short*)(ws + 67108864ull);
  unsigned short* wob = (unsigned short*)(ws + 83886080ull);
  float* xg  = (float*)(ws + 92274688ull);
  float* xc  = (float*)(ws + 360710144ull);
  float* ysb = (float*)(ws + 494927872ull);
  unsigned short* yb = (unsigned short*)(ws + 629145600ull);

  // 0) f32 -> bf16 operand conversion
  cvt_f32_bf16<<<(BS * DD + 255) / 256, 256, 0, stream>>>(x, xb, BS * DD);
  cvt_f32_bf16<<<(C2 * DD + 255) / 256, 256, 0, stream>>>(w_in, wb, C2 * DD);
  cvt_f32_bf16<<<(DD * CC + 255) / 256, 256, 0, stream>>>(w_out, wob, DD * CC);

  // 1) xg = x @ w_in^T   (16384 x 2048) * (4096 x 2048)^T
  gemm_bf16_wmma<<<dim3(BS / 128, C2 / 128), 256, 0, stream>>>(xb, wb, xg, BS, C2);

  // 2) xin -> causal depthwise conv + SiLU
  conv_silu_kernel<<<(int)(((long)BS * CC) / 256), 256, 0, stream>>>(xg, conv_w, xc);

  // 3) sequential scan: one wave per head
  rnn_scan_kernel<<<NN, 32, 0, stream>>>(xc, state_w, ysb);

  // 4) gate + RMSNorm -> bf16
  gate_norm_kernel<<<BS, 256, 0, stream>>>(ysb, xg, norm_w, yb);

  // 5) out = y @ w_out^T   (16384 x 2048) * (2048 x 2048)^T
  gemm_bf16_wmma<<<dim3(BS / 128, DD / 128), 256, 0, stream>>>(yb, wob, out, BS, DD);
}
